// GCNModel_50989851738541
// MI455X (gfx1250) — compile-verified, matrix operand order checked by reading, not run on hardware
//
#include <hip/hip_runtime.h>

// ---------------------------------------------------------------------------
// CDNA5 (gfx1250) WMMA types
// ---------------------------------------------------------------------------
typedef __attribute__((ext_vector_type(16))) __bf16 v16bf;
typedef __attribute__((ext_vector_type(8)))  float  v8f;

union frag_u { uint4 q[2]; v16bf v; };

// ---------------------------------------------------------------------------
// A-fragment (16x32 bf16, MxK), row-major bf16 source. ISA layout:
//   lanes 0-15 : row = lane,    elems 0-7 -> K=k0+0..7,  elems 8-15 -> K=k0+16..23
//   lanes16-31 : row = lane-16, elems 0-7 -> K=k0+8..15, elems 8-15 -> K=k0+24..31
// OOB rows are clamped (their accumulator rows are never stored).
// B-fragments come from a pre-packed layout: tile (kt,nt) = 32 lanes x 16 bf16
// contiguous (1KB), per-lane one 32B run, wave reads fully coalesced.
// ---------------------------------------------------------------------------

// Compile-time-shape GEMM: H = A(MxK,bf16) * Bpk(packed bf16), fp32 accum.
// 256 threads = 8 waves; wave tile 32x64 (2x4 WMMAs), block tile 128x128.
// All fragment loads are immediate-offset b128s from two base pointers.
template <int K, int N>
__global__ __launch_bounds__(256)
void gemm_wmma_t(const __bf16* __restrict__ A, const __bf16* __restrict__ Bpk,
                 float* __restrict__ H, int M) {
  constexpr int NT = N >> 4;                 // #B tiles along N
  const int wave = threadIdx.x >> 5;
  const int lane = threadIdx.x & 31;
  const int m0 = blockIdx.x * 128 + (wave & 3) * 32;
  const int n0 = blockIdx.y * 128 + (wave >> 2) * 64;

  int r0 = m0 + (lane & 15);      r0 = (r0 < M) ? r0 : (M - 1);
  int r1 = m0 + 16 + (lane & 15); r1 = (r1 < M) ? r1 : (M - 1);
  const int kbo = (lane >= 16) ? 8 : 0;
  const __bf16* pa0 = A + (size_t)r0 * K + kbo;
  const __bf16* pa1 = A + (size_t)r1 * K + kbo;
  const uint4*  pb  = (const uint4*)Bpk + ((size_t)(n0 >> 4) * 32 + lane) * 2;

  v8f c[2][4] = {};

#pragma unroll
  for (int kt = 0; kt < K / 32; ++kt) {
    frag_u a0, a1, b[4];
    const uint4* qa0 = (const uint4*)(pa0 + kt * 32);
    const uint4* qa1 = (const uint4*)(pa1 + kt * 32);
    a0.q[0] = qa0[0]; a0.q[1] = qa0[2];      // +0B, +32B (immediate offsets)
    a1.q[0] = qa1[0]; a1.q[1] = qa1[2];
#pragma unroll
    for (int ni = 0; ni < 4; ++ni) {
      const uint4* qb = pb + (size_t)kt * NT * 64 + ni * 64;   // const offsets
      b[ni].q[0] = qb[0]; b[ni].q[1] = qb[1];
    }

    c[0][0] = __builtin_amdgcn_wmma_f32_16x16x32_bf16(false, a0.v, false, b[0].v, (short)0, c[0][0], false, false);
    c[0][1] = __builtin_amdgcn_wmma_f32_16x16x32_bf16(false, a0.v, false, b[1].v, (short)0, c[0][1], false, false);
    c[0][2] = __builtin_amdgcn_wmma_f32_16x16x32_bf16(false, a0.v, false, b[2].v, (short)0, c[0][2], false, false);
    c[0][3] = __builtin_amdgcn_wmma_f32_16x16x32_bf16(false, a0.v, false, b[3].v, (short)0, c[0][3], false, false);
    c[1][0] = __builtin_amdgcn_wmma_f32_16x16x32_bf16(false, a1.v, false, b[0].v, (short)0, c[1][0], false, false);
    c[1][1] = __builtin_amdgcn_wmma_f32_16x16x32_bf16(false, a1.v, false, b[1].v, (short)0, c[1][1], false, false);
    c[1][2] = __builtin_amdgcn_wmma_f32_16x16x32_bf16(false, a1.v, false, b[2].v, (short)0, c[1][2], false, false);
    c[1][3] = __builtin_amdgcn_wmma_f32_16x16x32_bf16(false, a1.v, false, b[3].v, (short)0, c[1][3], false, false);
  }

  // C/D layout: VGPR r -> M = r (lanes 0-15) / r+8 (lanes 16-31), N = lane&15
  const int nl = lane & 15;
  const int mh = (lane >> 4) * 8;
#pragma unroll
  for (int mi = 0; mi < 2; ++mi)
#pragma unroll
    for (int ni = 0; ni < 4; ++ni)
#pragma unroll
      for (int r = 0; r < 8; ++r) {
        const int m = m0 + mi * 16 + mh + r;
        const int n = n0 + ni * 16 + nl;
        if (m < M) H[(size_t)m * N + n] = c[mi][ni][r];
      }
}

// Dynamic-shape fallback (same algorithm, runtime K/N).
__global__ __launch_bounds__(256)
void gemm_wmma_dyn(const __bf16* __restrict__ A, const __bf16* __restrict__ Bpk,
                   float* __restrict__ H, int M, int K, int N) {
  const int wave = threadIdx.x >> 5;
  const int lane = threadIdx.x & 31;
  const int m0 = blockIdx.x * 128 + (wave & 3) * 32;
  const int n0 = blockIdx.y * 128 + (wave >> 2) * 64;
  const int NT = N >> 4;

  int r0 = m0 + (lane & 15);      r0 = (r0 < M) ? r0 : (M - 1);
  int r1 = m0 + 16 + (lane & 15); r1 = (r1 < M) ? r1 : (M - 1);
  const int kbo = (lane >= 16) ? 8 : 0;
  const __bf16* pa0 = A + (size_t)r0 * K + kbo;
  const __bf16* pa1 = A + (size_t)r1 * K + kbo;
  const uint4*  pb  = (const uint4*)Bpk + ((size_t)(n0 >> 4) * 32 + lane) * 2;
  const size_t  bstep = (size_t)NT * 64;

  v8f c[2][4] = {};
  for (int kt = 0; kt < (K >> 5); ++kt) {
    frag_u a0, a1, b[4];
    const uint4* qa0 = (const uint4*)(pa0 + (size_t)kt * 32);
    const uint4* qa1 = (const uint4*)(pa1 + (size_t)kt * 32);
    a0.q[0] = qa0[0]; a0.q[1] = qa0[2];
    a1.q[0] = qa1[0]; a1.q[1] = qa1[2];
#pragma unroll
    for (int ni = 0; ni < 4; ++ni) {
      const uint4* qb = pb + ni * 64;
      b[ni].q[0] = qb[0]; b[ni].q[1] = qb[1];
    }
    pb += bstep;

    c[0][0] = __builtin_amdgcn_wmma_f32_16x16x32_bf16(false, a0.v, false, b[0].v, (short)0, c[0][0], false, false);
    c[0][1] = __builtin_amdgcn_wmma_f32_16x16x32_bf16(false, a0.v, false, b[1].v, (short)0, c[0][1], false, false);
    c[0][2] = __builtin_amdgcn_wmma_f32_16x16x32_bf16(false, a0.v, false, b[2].v, (short)0, c[0][2], false, false);
    c[0][3] = __builtin_amdgcn_wmma_f32_16x16x32_bf16(false, a0.v, false, b[3].v, (short)0, c[0][3], false, false);
    c[1][0] = __builtin_amdgcn_wmma_f32_16x16x32_bf16(false, a1.v, false, b[0].v, (short)0, c[1][0], false, false);
    c[1][1] = __builtin_amdgcn_wmma_f32_16x16x32_bf16(false, a1.v, false, b[1].v, (short)0, c[1][1], false, false);
    c[1][2] = __builtin_amdgcn_wmma_f32_16x16x32_bf16(false, a1.v, false, b[2].v, (short)0, c[1][2], false, false);
    c[1][3] = __builtin_amdgcn_wmma_f32_16x16x32_bf16(false, a1.v, false, b[3].v, (short)0, c[1][3], false, false);
  }

  const int nl = lane & 15;
  const int mh = (lane >> 4) * 8;
#pragma unroll
  for (int mi = 0; mi < 2; ++mi)
#pragma unroll
    for (int ni = 0; ni < 4; ++ni)
#pragma unroll
      for (int r = 0; r < 8; ++r) {
        const int m = m0 + mi * 16 + mh + r;
        const int n = n0 + ni * 16 + nl;
        if (m < M) H[(size_t)m * N + n] = c[mi][ni][r];
      }
}

// ---------------------------------------------------------------------------
// Pre-processing kernels
// ---------------------------------------------------------------------------

// fp32 -> bf16 row-major, 8 elements / thread (b128 in, b128 out)
__global__ void cvt_bf16_kernel(const float* __restrict__ src, __bf16* __restrict__ dst,
                                long long n8) {
  long long i = (long long)blockIdx.x * blockDim.x + threadIdx.x;
  if (i >= n8) return;
  const float4* p = (const float4*)src + i * 2;
  const float4 x = p[0], y = p[1];
  union { uint4 q; __bf16 b[8]; } o;
  o.b[0] = (__bf16)x.x; o.b[1] = (__bf16)x.y; o.b[2] = (__bf16)x.z; o.b[3] = (__bf16)x.w;
  o.b[4] = (__bf16)y.x; o.b[5] = (__bf16)y.y; o.b[6] = (__bf16)y.z; o.b[7] = (__bf16)y.w;
  ((uint4*)dst)[i] = o.q;
}

// Pack W (KxN fp32, row-major) into per-lane WMMA B-fragment layout (bf16).
__global__ void pack_w_kernel(const float* __restrict__ W, __bf16* __restrict__ Wpk,
                              int K, int N, long long total) {
  long long t = (long long)blockIdx.x * blockDim.x + threadIdx.x;
  if (t >= total) return;
  const int ntiles = N >> 4;
  const int lane = (int)(t & 31);
  const long long tile = t >> 5;
  const int nt = (int)(tile % ntiles);
  const int kt = (int)(tile / ntiles);
  const int n  = (nt << 4) + (lane & 15);
  const int kb = (kt << 5) + ((lane >= 16) ? 8 : 0);
  union { uint4 q[2]; __bf16 b[16]; } o;
#pragma unroll
  for (int i = 0; i < 8; ++i) o.b[i]     = (__bf16)W[(size_t)(kb + i)      * N + n];
#pragma unroll
  for (int i = 0; i < 8; ++i) o.b[8 + i] = (__bf16)W[(size_t)(kb + 16 + i) * N + n];
  uint4* dst = (uint4*)Wpk + t * 2;
  dst[0] = o.q[0];
  dst[1] = o.q[1];
}

// ---------------------------------------------------------------------------
// Graph / elementwise kernels
// ---------------------------------------------------------------------------
__global__ void fill_kernel(float* __restrict__ p, float v, long long n) {
  long long i = (long long)blockIdx.x * blockDim.x + threadIdx.x;
  if (i < n) p[i] = v;
}

__global__ void deg_kernel(const int* __restrict__ col, float* __restrict__ deg, int E) {
  int e = blockIdx.x * blockDim.x + threadIdx.x;
  if (e < E) atomicAdd(&deg[col[e]], 1.0f);
}

__global__ void rsqrt_kernel(const float* __restrict__ deg, float* __restrict__ dinv, int M) {
  int i = blockIdx.x * blockDim.x + threadIdx.x;
  if (i < M) dinv[i] = rsqrtf(deg[i]);   // deg >= 1 always (self-loop)
}

// AGG[col] += H[row] * (dinv[row]*dinv[col]); 4 floats per thread, feature fastest.
__global__ void scatter_kernel(const float* __restrict__ H, const int* __restrict__ row,
                               const int* __restrict__ col, const float* __restrict__ dinv,
                               float* __restrict__ AGG, long long total, int vshift, int cshift) {
  long long tid = (long long)blockIdx.x * blockDim.x + threadIdx.x;
  if (tid >= total) return;
  const long long e = tid >> vshift;
  const int f = (int)(tid & ((1 << vshift) - 1)) << 2;
  const int r = row[e], c = col[e];
  const float nrm = dinv[r] * dinv[c];
  const float4 h = *(const float4*)(H + ((size_t)r << cshift) + f);
  float* dst = AGG + ((size_t)c << cshift) + f;
  atomicAdd(dst + 0, h.x * nrm);
  atomicAdd(dst + 1, h.y * nrm);
  atomicAdd(dst + 2, h.z * nrm);
  atomicAdd(dst + 3, h.w * nrm);
}

// out = dropout(relu(AGG + H*dinv^2 + bias)); T = __bf16 (layer-1 act) or float
template <typename T>
__global__ void combine_kernel(const float* __restrict__ AGG, const float* __restrict__ H,
                               const float* __restrict__ dinv, const float* __restrict__ bias,
                               T* __restrict__ out, long long total, int cshift, unsigned seed) {
  long long tid = (long long)blockIdx.x * blockDim.x + threadIdx.x;
  if (tid >= total) return;
  const long long i = tid >> cshift;
  const int f = (int)(tid & ((1 << cshift) - 1));
  const float d = dinv[i];
  float v = AGG[tid] + H[tid] * d * d + bias[f];
  v = fmaxf(v, 0.0f);
  unsigned h = (unsigned)tid * 2654435761u + seed;
  h ^= h >> 16; h *= 0x7feb352du; h ^= h >> 15; h *= 0x846ca68bu; h ^= h >> 16;
  out[tid] = (T)((h & 1u) ? (v + v) : 0.0f);   // keep-prob 0.5, scale 1/0.5
}

// ---------------------------------------------------------------------------
// Host-side orchestration
// ---------------------------------------------------------------------------
static inline long long cdiv_ll(long long a, long long b) { return (a + b - 1) / b; }

static void launch_gemm(const __bf16* A, const __bf16* Bpk, float* H,
                        int M, int K, int N, hipStream_t stream) {
  dim3 grid((unsigned)cdiv_ll(M, 128), (unsigned)(N / 128));
  if (K == 256 && N == 512)
    gemm_wmma_t<256, 512><<<grid, 256, 0, stream>>>(A, Bpk, H, M);
  else if (K == 512 && N == 256)
    gemm_wmma_t<512, 256><<<grid, 256, 0, stream>>>(A, Bpk, H, M);
  else
    gemm_wmma_dyn<<<grid, 256, 0, stream>>>(A, Bpk, H, M, K, N);
}

extern "C" void kernel_launch(void* const* d_in, const int* in_sizes, int n_in,
                              void* d_out, int out_size, void* d_ws, size_t ws_size,
                              hipStream_t stream) {
  const float* X  = (const float*)d_in[0];
  const int*   ei = (const int*)d_in[1];
  const float* W1 = (const float*)d_in[2];
  const float* b1 = (const float*)d_in[3];
  const float* W2 = (const float*)d_in[4];
  const float* b2 = (const float*)d_in[5];

  const int HID = in_sizes[3];                 // 512
  const int OUT = in_sizes[5];                 // 256
  const int IN  = in_sizes[2] / HID;           // 256
  const int M   = in_sizes[0] / IN;            // 50000
  const int E   = in_sizes[1] / 2;             // 800000
  const int* row = ei;
  const int* col = ei + E;

  const int hshift = __builtin_ctz((unsigned)HID);   // 9
  const int oshift = __builtin_ctz((unsigned)OUT);   // 8

  // Workspace layout:
  //   H1   : M*HID f32   (layer-2 reuse: H2 = first M*OUT, AGG2 = next M*OUT)
  //   AGG1 : M*HID f32
  //   deg  : M f32,  dinv : M f32
  //   Xbf  : M*IN  bf16, X2bf : M*HID bf16
  //   W1pk : IN*HID bf16, W2pk : HID*OUT bf16 (packed B fragments)
  char* ws = (char*)d_ws;
  float*  H1   = (float*)ws;                          ws += (size_t)M * HID * 4;
  float*  AGG1 = (float*)ws;                          ws += (size_t)M * HID * 4;
  float*  deg  = (float*)ws;                          ws += (size_t)M * 4;
  float*  dinv = (float*)ws;                          ws += (size_t)M * 4;
  __bf16* Xbf  = (__bf16*)ws;                         ws += (size_t)M * IN * 2;
  __bf16* X2bf = (__bf16*)ws;                         ws += (size_t)M * HID * 2;
  __bf16* W1pk = (__bf16*)ws;                         ws += (size_t)IN * HID * 2;
  __bf16* W2pk = (__bf16*)ws;
  float*  H2   = H1;
  float*  AGG2 = H1 + (size_t)M * OUT;

  const int B = 256;

  // --- degrees / normalization -------------------------------------------
  fill_kernel<<<(unsigned)cdiv_ll(M, B), B, 0, stream>>>(deg, 1.0f, M);
  deg_kernel<<<(unsigned)cdiv_ll(E, B), B, 0, stream>>>(col, deg, E);
  rsqrt_kernel<<<(unsigned)cdiv_ll(M, B), B, 0, stream>>>(deg, dinv, M);

  // --- precision prep -----------------------------------------------------
  {
    const long long n8 = (long long)M * IN / 8;
    cvt_bf16_kernel<<<(unsigned)cdiv_ll(n8, B), B, 0, stream>>>(X, Xbf, n8);
    const long long t1 = (long long)(IN / 32) * (HID / 16) * 32;
    pack_w_kernel<<<(unsigned)cdiv_ll(t1, B), B, 0, stream>>>(W1, W1pk, IN, HID, t1);
    const long long t2 = (long long)(HID / 32) * (OUT / 16) * 32;
    pack_w_kernel<<<(unsigned)cdiv_ll(t2, B), B, 0, stream>>>(W2, W2pk, HID, OUT, t2);
  }

  // --- layer 1 ------------------------------------------------------------
  {
    const long long mc = (long long)M * HID;
    fill_kernel<<<(unsigned)cdiv_ll(mc, B), B, 0, stream>>>(AGG1, 0.0f, mc);

    launch_gemm(Xbf, W1pk, H1, M, IN, HID, stream);

    const int vshift = hshift - 2;
    const long long total = (long long)E << vshift;
    scatter_kernel<<<(unsigned)cdiv_ll(total, B), B, 0, stream>>>(
        H1, row, col, dinv, AGG1, total, vshift, hshift);

    combine_kernel<__bf16><<<(unsigned)cdiv_ll(mc, B), B, 0, stream>>>(
        AGG1, H1, dinv, b1, X2bf, mc, hshift, 0x9E3779B9u);
  }

  // --- layer 2 ------------------------------------------------------------
  {
    const long long mc = (long long)M * OUT;
    fill_kernel<<<(unsigned)cdiv_ll(mc, B), B, 0, stream>>>(AGG2, 0.0f, mc);

    launch_gemm(X2bf, W2pk, H2, M, HID, OUT, stream);

    const int vshift = oshift - 2;
    const long long total = (long long)E << vshift;
    scatter_kernel<<<(unsigned)cdiv_ll(total, B), B, 0, stream>>>(
        H2, row, col, dinv, AGG2, total, vshift, oshift);

    combine_kernel<float><<<(unsigned)cdiv_ll(mc, B), B, 0, stream>>>(
        AGG2, H2, dinv, b2, (float*)d_out, mc, oshift, 0x85EBCA6Bu);
  }
}
